// MultiHeadSelfAttention_89412629168654
// MI455X (gfx1250) — matrix operand order repaired and checked by reading, not compiled, not run on hardware
//
#include <hip/hip_runtime.h>

// MultiHeadSelfAttention for MI455X (gfx1250, wave32, WMMA).
// B=2, S=2048, D=1024, H=16, HD=64.
// bf16 data / f32 accumulate via v_wmma_f32_16x16x32_bf16.
// GEMMs: 64x64 tile per wave (16 WMMA per 16 b128 loads per K-step).
// Attention: flash-style, 32-query tile per wave, K/V staged through LDS
// via async global->LDS DMA (ASYNCcnt) when available, online softmax.

#define Bq 2
#define Sq 2048
#define Dq 1024
#define Hq 16
#define HDq 64

typedef __attribute__((ext_vector_type(16))) __bf16 v16bf;
typedef __attribute__((ext_vector_type(8)))  float  v8f;
typedef __attribute__((ext_vector_type(4)))  int    v4i_t;

union Frag16 { v16bf v; uint4 q[2]; };

#if defined(__has_builtin)
#if __has_builtin(__builtin_amdgcn_global_load_async_to_lds_b128) && \
    __has_builtin(__builtin_amdgcn_s_wait_asynccnt)
#define HAVE_ASYNC_LDS 1
#endif
#endif
#ifndef HAVE_ASYNC_LDS
#define HAVE_ASYNC_LDS 0
#endif

#if HAVE_ASYNC_LDS
typedef __attribute__((address_space(1))) v4i_t* gv4p;   // global v4i*
typedef __attribute__((address_space(3))) v4i_t* lv4p;   // LDS v4i*
#endif

__device__ __forceinline__ unsigned short f2bf(float f) {
    unsigned int u = __float_as_uint(f);
    u += 0x7FFFu + ((u >> 16) & 1u);       // round to nearest even
    return (unsigned short)(u >> 16);
}

// A-fragment (16x32 bf16, MxK): lane l holds row M=l%16.
// lanes 0-15: elems0-7 = K 0..7,  elems8-15 = K 16..23
// lanes16-31: elems0-7 = K 8..15, elems8-15 = K 24..31
__device__ __forceinline__ v16bf load_frag_a(const unsigned short* base, int ld, int lane) {
    const int m   = lane & 15;
    const int klo = (lane < 16) ? 0 : 8;
    const unsigned short* p = base + m * ld + klo;
    Frag16 f;
    f.q[0] = *(const uint4*)(p);
    f.q[1] = *(const uint4*)(p + 16);
    return f.v;
}

// B-fragment (32x16 bf16, KxN), source "N-major" (row n contiguous in K):
// lane l holds column N=l%16; lanes0-15 K=kbase+0..7,16..23, lanes16-31 K=kbase+8..15,24..31.
__device__ __forceinline__ v16bf load_frag_b(const unsigned short* base, int ld,
                                             int kbase, int lane) {
    const int n    = lane & 15;
    const int koff = kbase + ((lane < 16) ? 0 : 8);
    const unsigned short* p = base + n * ld + koff;
    Frag16 f;
    f.q[0] = *(const uint4*)(p);
    f.q[1] = *(const uint4*)(p + 16);
    return f.v;
}

// ---------------- conversion kernels ----------------

__global__ void cvt_bf16_kernel(const float* __restrict__ src,
                                unsigned short* __restrict__ dst, int n) {
    int i = blockIdx.x * blockDim.x + threadIdx.x;
    if (i < n) dst[i] = f2bf(src[i]);
}

// dst[n*D + k] = bf16(src[k*D + n])
__global__ void transpose_w_kernel(const float* __restrict__ src,
                                   unsigned short* __restrict__ dst) {
    int k = blockIdx.x * 16 + threadIdx.x;
    int n = blockIdx.y * 16 + threadIdx.y;
    dst[(size_t)n * Dq + k] = f2bf(src[(size_t)k * Dq + n]);
}

// ---------------- QKV projection GEMM ----------------
// One wave computes a 64x64 tile of X(4096x1024) @ W + b.
// Q,K -> [B,H,S,HD] bf16 ; V -> transposed [B,H,HD,S] bf16.
__global__ __launch_bounds__(32)
void gemm_qkv_kernel(const unsigned short* __restrict__ Xbf,
                     const unsigned short* __restrict__ WqT,
                     const unsigned short* __restrict__ WkT,
                     const unsigned short* __restrict__ WvT,
                     const float* __restrict__ bqp,
                     const float* __restrict__ bkp,
                     const float* __restrict__ bvp,
                     unsigned short* __restrict__ Qd,
                     unsigned short* __restrict__ Kd,
                     unsigned short* __restrict__ Vt) {
    const int lane  = threadIdx.x;
    const int m0    = blockIdx.x * 64;
    const int n0    = blockIdx.y * 64;
    const int which = blockIdx.z;
    const unsigned short* WT  = (which == 0) ? WqT : (which == 1) ? WkT : WvT;
    const float*          bia = (which == 0) ? bqp : (which == 1) ? bkp : bvp;

    v8f acc[4][4] = {};                       // [m-subtile][n-subtile]
    for (int k0 = 0; k0 < Dq; k0 += 32) {
        v16bf a[4];
#pragma unroll
        for (int mi = 0; mi < 4; ++mi)
            a[mi] = load_frag_a(Xbf + (size_t)(m0 + 16 * mi) * Dq + k0, Dq, lane);
#pragma unroll
        for (int nt = 0; nt < 4; ++nt) {
            v16bf b = load_frag_b(WT + (size_t)(n0 + 16 * nt) * Dq, Dq, k0, lane);
#pragma unroll
            for (int mi = 0; mi < 4; ++mi)
                acc[mi][nt] = __builtin_amdgcn_wmma_f32_16x16x32_bf16(
                    false, a[mi], false, b, (short)0, acc[mi][nt], false, false);
        }
    }

    const int mhalf = (lane & 16) ? 8 : 0;
#pragma unroll
    for (int nt = 0; nt < 4; ++nt) {
        const int n = n0 + 16 * nt + (lane & 15);
        const float bv_ = bia[n];
        const int h  = n >> 6;
        const int hd = n & 63;
#pragma unroll
        for (int mi = 0; mi < 4; ++mi) {
#pragma unroll
            for (int v = 0; v < 8; ++v) {
                const int m  = m0 + 16 * mi + mhalf + v;   // row in [0, B*S)
                const int bb = m >> 11;
                const int s  = m & 2047;
                const unsigned short val = f2bf(acc[mi][nt][v] + bv_);
                const size_t bh = (size_t)(bb * Hq + h);
                if (which == 0)      Qd[(bh * Sq + s) * HDq + hd] = val;
                else if (which == 1) Kd[(bh * Sq + s) * HDq + hd] = val;
                else                 Vt[(bh * HDq + hd) * Sq + s] = val;
            }
        }
    }
}

// ---------------- flash attention ----------------
// One wave per (b, h, 32-query tile). Keys streamed in blocks of 32.
// K tile (32x64) and V^T tile (64x32) staged in LDS (async DMA if available),
// scores via 8 WMMA, online softmax, P relayout through LDS, P.V via 8 WMMA.
__global__ __launch_bounds__(32)
void attn_kernel(const unsigned short* __restrict__ Qd,
                 const unsigned short* __restrict__ Kd,
                 const unsigned short* __restrict__ Vt,
                 unsigned short* __restrict__ Ab) {
    __shared__ __align__(16) unsigned short ldsK[32 * 64];   // [key][hd]
    __shared__ __align__(16) unsigned short ldsV[64 * 32];   // [hd][key]
    __shared__ __align__(16) unsigned short ldsP[32 * 32];   // [q][key]

    const int lane = threadIdx.x;
    const int q0   = blockIdx.x * 32;
    const int h    = blockIdx.y;
    const int b    = blockIdx.z;
    const size_t bh = (size_t)(b * Hq + h);
    const unsigned short* Qb = Qd + bh * Sq * HDq;
    const unsigned short* Kb = Kd + bh * Sq * HDq;
    const unsigned short* Vb = Vt + bh * HDq * Sq;

    v16bf aQ[2][2];
#pragma unroll
    for (int qt = 0; qt < 2; ++qt) {
        aQ[qt][0] = load_frag_a(Qb + (size_t)(q0 + 16 * qt) * HDq + 0,  HDq, lane);
        aQ[qt][1] = load_frag_a(Qb + (size_t)(q0 + 16 * qt) * HDq + 32, HDq, lane);
    }

    v8f O[2][4] = {};
    float mrow[2][8], lrow[2][8];
#pragma unroll
    for (int qt = 0; qt < 2; ++qt)
#pragma unroll
        for (int v = 0; v < 8; ++v) { mrow[qt][v] = -__builtin_inff(); lrow[qt][v] = 0.f; }

    const int mhalf = (lane & 16) ? 8 : 0;
    const int col   = lane & 15;

    for (int k0 = 0; k0 < Sq; k0 += 32) {
        // ---- stage K (32x64, contiguous 4KB) and V^T (64 rows x 32 keys) in LDS
        const unsigned short* gK = Kb + (size_t)k0 * HDq;
#if HAVE_ASYNC_LDS
#pragma unroll
        for (int i = 0; i < 8; ++i) {
            const int c = i * 32 + lane;                       // 16B chunk id
            __builtin_amdgcn_global_load_async_to_lds_b128(
                (gv4p)(gK + c * 8), (lv4p)(ldsK + c * 8), 0, 0);
            const int row = c >> 2, off = (c & 3) * 8;
            __builtin_amdgcn_global_load_async_to_lds_b128(
                (gv4p)(Vb + (size_t)row * Sq + k0 + off),
                (lv4p)(ldsV + row * 32 + off), 0, 0);
        }
        __builtin_amdgcn_s_wait_asynccnt(0);
#else
#pragma unroll
        for (int i = 0; i < 8; ++i) {
            const int c = i * 32 + lane;
            *(uint4*)(ldsK + c * 8) = *(const uint4*)(gK + c * 8);
            const int row = c >> 2, off = (c & 3) * 8;
            *(uint4*)(ldsV + row * 32 + off) =
                *(const uint4*)(Vb + (size_t)row * Sq + k0 + off);
        }
#endif
        __syncthreads();

        // ---- scores: S[qt][kt] = Q[qt] . K[kt]^T  (contraction over HD=64)
        v8f s[2][2];
#pragma unroll
        for (int kt = 0; kt < 2; ++kt) {
            v16bf bK0 = load_frag_b(ldsK + 16 * kt * 64, 64, 0,  lane);
            v16bf bK1 = load_frag_b(ldsK + 16 * kt * 64, 64, 32, lane);
#pragma unroll
            for (int qt = 0; qt < 2; ++qt) {
                v8f t = {};
                t = __builtin_amdgcn_wmma_f32_16x16x32_bf16(
                    false, aQ[qt][0], false, bK0, (short)0, t, false, false);
                t = __builtin_amdgcn_wmma_f32_16x16x32_bf16(
                    false, aQ[qt][1], false, bK1, (short)0, t, false, false);
                s[qt][kt] = t;
            }
        }

        // ---- online softmax per row (row = one VGPR across a 16-lane half)
#pragma unroll
        for (int qt = 0; qt < 2; ++qt) {
#pragma unroll
            for (int v = 0; v < 8; ++v) {
                float x0 = s[qt][0][v] * 0.125f;   // 1/sqrt(HD)
                float x1 = s[qt][1][v] * 0.125f;
                float mx = fmaxf(x0, x1);
                mx = fmaxf(mx, __shfl_xor(mx, 1));
                mx = fmaxf(mx, __shfl_xor(mx, 2));
                mx = fmaxf(mx, __shfl_xor(mx, 4));
                mx = fmaxf(mx, __shfl_xor(mx, 8));
                const float mnew  = fmaxf(mrow[qt][v], mx);
                const float alpha = __expf(mrow[qt][v] - mnew);
                mrow[qt][v] = mnew;
                const float p0 = __expf(x0 - mnew);
                const float p1 = __expf(x1 - mnew);
                float rs = p0 + p1;
                rs += __shfl_xor(rs, 1);
                rs += __shfl_xor(rs, 2);
                rs += __shfl_xor(rs, 4);
                rs += __shfl_xor(rs, 8);
                lrow[qt][v] = lrow[qt][v] * alpha + rs;
                O[qt][0][v] *= alpha; O[qt][1][v] *= alpha;
                O[qt][2][v] *= alpha; O[qt][3][v] *= alpha;
                const int M = 16 * qt + mhalf + v;
                ldsP[M * 32 + col]      = f2bf(p0);
                ldsP[M * 32 + col + 16] = f2bf(p1);
            }
        }
        __syncthreads();

        // ---- P.V : A = P rows from LDS, B = V^T rows (keys contiguous)
#pragma unroll
        for (int qt = 0; qt < 2; ++qt) {
            const v16bf aP = load_frag_a(ldsP + 16 * qt * 32, 32, lane);
#pragma unroll
            for (int n = 0; n < 4; ++n) {
                v16bf bV = load_frag_b(ldsV + 16 * n * 32, 32, 0, lane);
                O[qt][n] = __builtin_amdgcn_wmma_f32_16x16x32_bf16(
                    false, aP, false, bV, (short)0, O[qt][n], false, false);
            }
        }
        __syncthreads();   // protect LDS WAR before next key block
    }

    // ---- normalize and store to [B,S,H*HD] bf16 buffer
#pragma unroll
    for (int qt = 0; qt < 2; ++qt) {
#pragma unroll
        for (int n = 0; n < 4; ++n) {
            const int c = h * HDq + 16 * n + col;
#pragma unroll
            for (int v = 0; v < 8; ++v) {
                const int q = q0 + 16 * qt + mhalf + v;
                Ab[((size_t)(b * Sq + q)) * Dq + c] = f2bf(O[qt][n][v] / lrow[qt][v]);
            }
        }
    }
}

// ---------------- output projection GEMM (64x64 tile per wave) ----------------
__global__ __launch_bounds__(32)
void gemm_out_kernel(const unsigned short* __restrict__ Ab,
                     const unsigned short* __restrict__ WoT,
                     const float* __restrict__ bop,
                     float* __restrict__ out) {
    const int lane = threadIdx.x;
    const int m0   = blockIdx.x * 64;
    const int n0   = blockIdx.y * 64;

    v8f acc[4][4] = {};
    for (int k0 = 0; k0 < Dq; k0 += 32) {
        v16bf a[4];
#pragma unroll
        for (int mi = 0; mi < 4; ++mi)
            a[mi] = load_frag_a(Ab + (size_t)(m0 + 16 * mi) * Dq + k0, Dq, lane);
#pragma unroll
        for (int nt = 0; nt < 4; ++nt) {
            v16bf b = load_frag_b(WoT + (size_t)(n0 + 16 * nt) * Dq, Dq, k0, lane);
#pragma unroll
            for (int mi = 0; mi < 4; ++mi)
                acc[mi][nt] = __builtin_amdgcn_wmma_f32_16x16x32_bf16(
                    false, a[mi], false, b, (short)0, acc[mi][nt], false, false);
        }
    }

    const int mhalf = (lane & 16) ? 8 : 0;
#pragma unroll
    for (int nt = 0; nt < 4; ++nt) {
        const int n = n0 + 16 * nt + (lane & 15);
        const float bv_ = bop[n];
#pragma unroll
        for (int mi = 0; mi < 4; ++mi) {
#pragma unroll
            for (int v = 0; v < 8; ++v) {
                const int m = m0 + 16 * mi + mhalf + v;
                out[(size_t)m * Dq + n] = acc[mi][nt][v] + bv_;
            }
        }
    }
}

extern "C" void kernel_launch(void* const* d_in, const int* in_sizes, int n_in,
                              void* d_out, int out_size, void* d_ws, size_t ws_size,
                              hipStream_t stream) {
    const float* X  = (const float*)d_in[0];
    const float* Wq = (const float*)d_in[1];
    const float* bq = (const float*)d_in[2];
    const float* Wk = (const float*)d_in[3];
    const float* bk = (const float*)d_in[4];
    const float* Wv = (const float*)d_in[5];
    const float* bv = (const float*)d_in[6];
    const float* Wo = (const float*)d_in[7];
    const float* bo = (const float*)d_in[8];
    float* out = (float*)d_out;

    // Workspace layout (48 MB total)
    char* ws = (char*)d_ws;
    unsigned short* Xbf = (unsigned short*)(ws);                  //  8 MB [4096,1024]
    unsigned short* WqT = (unsigned short*)(ws + ( 8ull << 20));  //  2 MB each
    unsigned short* WkT = (unsigned short*)(ws + (10ull << 20));
    unsigned short* WvT = (unsigned short*)(ws + (12ull << 20));
    unsigned short* WoT = (unsigned short*)(ws + (14ull << 20));
    unsigned short* Qd  = (unsigned short*)(ws + (16ull << 20));  //  8 MB [B,H,S,HD]
    unsigned short* Kd  = (unsigned short*)(ws + (24ull << 20));  //  8 MB [B,H,S,HD]
    unsigned short* Vt  = (unsigned short*)(ws + (32ull << 20));  //  8 MB [B,H,HD,S]
    unsigned short* Ab  = (unsigned short*)(ws + (40ull << 20));  //  8 MB [B,S,D]

    const int nX = Bq * Sq * Dq;
    cvt_bf16_kernel<<<nX / 256, 256, 0, stream>>>(X, Xbf, nX);

    dim3 tb(16, 16);
    dim3 tg(Dq / 16, Dq / 16);
    transpose_w_kernel<<<tg, tb, 0, stream>>>(Wq, WqT);
    transpose_w_kernel<<<tg, tb, 0, stream>>>(Wk, WkT);
    transpose_w_kernel<<<tg, tb, 0, stream>>>(Wv, WvT);
    transpose_w_kernel<<<tg, tb, 0, stream>>>(Wo, WoT);

    gemm_qkv_kernel<<<dim3((Bq * Sq) / 64, Dq / 64, 3), 32, 0, stream>>>(
        Xbf, WqT, WkT, WvT, bq, bk, bv, Qd, Kd, Vt);

    attn_kernel<<<dim3(Sq / 32, Hq, Bq), 32, 0, stream>>>(Qd, Kd, Vt, Ab);

    gemm_out_kernel<<<dim3((Bq * Sq) / 64, Dq / 64), 32, 0, stream>>>(Ab, WoT, bo, out);
}
